// LinearCrossEntropy_88381837017243
// MI455X (gfx1250) — matrix operand order repaired and checked by reading, not compiled, not run on hardware
//
#include <hip/hip_runtime.h>
#include <cstdint>
#include <cstddef>

// CDNA5 wave32 WMMA fragment/vector types
typedef __attribute__((ext_vector_type(16))) __bf16 v16bf;
typedef __attribute__((ext_vector_type(8)))  float  v8f;

// TDM descriptor SGPR-group types
typedef unsigned int u32x4 __attribute__((ext_vector_type(4)));
typedef int          i32x4 __attribute__((ext_vector_type(4)));
typedef int          i32x8 __attribute__((ext_vector_type(8)));

namespace {

constexpr int kD      = 1024;   // hidden dim
constexpr int kV      = 50257;  // vocab
constexpr int kN      = 8192;   // rows (B*T)
constexpr int kBlock  = 256;    // 8 waves of 32
constexpr int kWaves  = 8;
constexpr int MROWS   = 32;     // rows per workgroup (2 WMMA M-tiles)
constexpr int MTILES  = 2;
constexpr int NCOLS   = 128;    // vocab columns per iteration (8 waves * 16)
constexpr int KCHUNK  = 128;    // K staged per TDM transfer
constexpr int KSTEPS  = kD / KCHUNK;
constexpr int CPAD    = 136;    // halves per LDS row: 128 data + 8 pad (bank spread)
constexpr int IGNORE  = -100;

// LDS layout (dynamic shared assumed at group-segment offset 0):
//   [0)              c_f32 double buffer: 2 x (128 x 128 f32) = 131072 B (TDM dest)
//   [131072)         e_hi / e_lo / c_hi / c_lo bf16 buffers + softmax stats
constexpr int CF32_FLOATS  = NCOLS * KCHUNK;          // one buffer
constexpr int CF32_BYTES   = CF32_FLOATS * 4;         // 65536
constexpr int BF_BASE      = 2 * CF32_BYTES;          // 131072

constexpr int E_HALVES = MROWS * CPAD;
constexpr int C_HALVES = NCOLS * CPAD;

constexpr size_t LDS_BYTES =
    (size_t)BF_BASE
  + (size_t)(2 * E_HALVES + 2 * C_HALVES) * 2
  + (size_t)(2 * kWaves * MROWS) * 4
  + (size_t)MROWS * 4
  + (size_t)MROWS * 4;

union FragV {                 // one WMMA A/B operand: 16 bf16 = 2 x b128 LDS loads
  v16bf v;
  uint4 q[2];
};

union Pack4 {                 // 4 bf16 packed for one b64 LDS store
  __bf16 h[4];
  uint2  u;
};

__device__ inline void split4(const float4& x, Pack4& hi, Pack4& lo) {
  float xs[4] = {x.x, x.y, x.z, x.w};
#pragma unroll
  for (int j = 0; j < 4; ++j) {
    __bf16 h = (__bf16)xs[j];                 // RNE truncation to bf16
    hi.h[j] = h;
    lo.h[j] = (__bf16)(xs[j] - (float)h);     // residual (bf16x3 emulation)
  }
}

// Issue one TDM transfer: 2D tile (rows x 128 f32), row stride kD f32,
// from global `gptr` into LDS byte offset `lds_off`. Rows beyond
// tensor_dim1=rows read as zero (handles the final partial vocab chunk).
__device__ inline void tdm_load_c(const float* gptr, unsigned int lds_off, int rows) {
  unsigned long long ga = (unsigned long long)(uintptr_t)gptr;
  u32x4 g0;
  g0[0] = 1u;                                         // count=1 (valid), user mode
  g0[1] = lds_off;                                    // lds_addr (bytes)
  g0[2] = (unsigned int)ga;                           // global_addr[31:0]
  g0[3] = (unsigned int)((ga >> 32) & 0x01FFFFFFu)    // global_addr[56:32]
        | (2u << 30);                                 // type=2 ("image")
  i32x8 g1;
  g1[0] = (int)(2u << 16);            // data_size=2 -> 4-byte elements; no flags
  g1[1] = (int)((unsigned)KCHUNK << 16);              // tensor_dim0[15:0]=128
  g1[2] = (int)(((unsigned)rows & 0xFFFFu) << 16);    // dim0 hi=0 | tensor_dim1=rows
  g1[3] = (int)((unsigned)KCHUNK << 16);              // dim1 hi=0 | tile_dim0=128
  g1[4] = NCOLS;                                      // tile_dim1=128 | tile_dim2=0
  g1[5] = kD;                                         // tensor_dim0_stride[31:0]=1024
  g1[6] = 0;                                          // stride hi | tensor_dim1_stride lo
  g1[7] = 0;
  i32x4 z4 = {0, 0, 0, 0};
#if defined(__clang_major__) && (__clang_major__ >= 23)
  i32x8 z8 = {0, 0, 0, 0, 0, 0, 0, 0};
  __builtin_amdgcn_tensor_load_to_lds(g0, g1, z4, z4, z8, 0);
#else
  __builtin_amdgcn_tensor_load_to_lds(g0, g1, z4, z4, 0);
#endif
}

} // namespace

__global__ void lce_init_kernel(float* g) { g[0] = 0.f; g[1] = 0.f; }

__global__ void lce_fin_kernel(const float* g, float* out) {
  out[0] = g[0] / fmaxf(g[1], 1.0f);
}

__global__ __launch_bounds__(kBlock)
void lce_main_kernel(const float* __restrict__ e,
                     const float* __restrict__ cmat,
                     const int*   __restrict__ targets,
                     const float* __restrict__ bias,
                     float*       __restrict__ gacc) {
  extern __shared__ __align__(16) char smem[];
  float*  c_f32   = (float*)smem;                       // 2 x 16384 f32 (TDM dest)
  __bf16* e_hi    = (__bf16*)(smem + BF_BASE);
  __bf16* e_lo    = e_hi + E_HALVES;
  __bf16* c_hi    = e_lo + E_HALVES;
  __bf16* c_lo    = c_hi + C_HALVES;
  float*  M_ws    = (float*)(c_lo + C_HALVES);
  float*  S_ws    = M_ws + kWaves * MROWS;
  float*  t_logit = S_ws + kWaves * MROWS;
  int*    tgt_lds = (int*)(t_logit + MROWS);

  const int tid    = threadIdx.x;
  const int lane   = tid & 31;
  const int w      = tid >> 5;       // wave id 0..7 -> 16-col vocab stripe
  const int lane15 = lane & 15;
  const int kh     = lane >> 4;      // K-half select (A) / row-group (C)
  const int row0   = blockIdx.x * MROWS;

  if (tid < MROWS) {
    t_logit[tid] = 0.f;
    tgt_lds[tid] = targets[row0 + tid];
  }
  __syncthreads();

  int   tgt_reg[MTILES][8];
  float m_run[MTILES][8], s_run[MTILES][8];
#pragma unroll
  for (int mt = 0; mt < MTILES; ++mt)
#pragma unroll
    for (int r = 0; r < 8; ++r) {
      tgt_reg[mt][r] = tgt_lds[mt * 16 + kh * 8 + r];
      m_run[mt][r]   = -__builtin_inff();
      s_run[mt][r]   = 0.f;
    }

  for (int v0 = 0; v0 < kV; v0 += NCOLS) {
    int rows = kV - v0;
    if (rows > NCOLS) rows = NCOLS;

    v8f acc[MTILES];
    v8f zero = {0.f, 0.f, 0.f, 0.f, 0.f, 0.f, 0.f, 0.f};
    acc[0] = zero;
    acc[1] = zero;

    // Speculative prefetch of the next vocab chunk (global_prefetch_b8).
    {
      int vn = v0 + NCOLS + (tid & 127);
      if (vn >= kV) vn = kV - 1;
      __builtin_prefetch((const void*)(cmat + (size_t)vn * kD), 0, 1);
    }

    // Kick off the TDM for chunk 0 of this vocab stripe (wave 0 only; the
    // previous reader of buffer 0 finished strictly before a barrier wave 0
    // has already passed).
    if (tid < 32) tdm_load_c(cmat + (size_t)v0 * kD, 0u, rows);

    for (int k = 0; k < KSTEPS; ++k) {
      const int k0 = k * KCHUNK;

      if (tid < 32) {
        if (k + 1 < KSTEPS) {
          // Issue next chunk into the other buffer, then wait for the older
          // in-flight TDM (current chunk) -> tensorcnt <= 1.
          tdm_load_c(cmat + (size_t)v0 * kD + (k0 + KCHUNK),
                     (unsigned int)(((k + 1) & 1) * CF32_BYTES), rows);
          __builtin_amdgcn_s_wait_tensorcnt(1);
        } else {
          __builtin_amdgcn_s_wait_tensorcnt(0);
        }
      }
      __syncthreads();   // TDM data visible; previous compute done with bf16 bufs

      // ---- stage e chunk: 32 rows x 128 K, f32 -> bf16 hi/lo into LDS ----
#pragma unroll
      for (int i = 0; i < 4; ++i) {
        int idx = tid + i * kBlock;          // 0..1023
        int rr  = idx >> 5;                  // row 0..31
        int q   = idx & 31;                  // float4 within row
        float4 x = *(const float4*)(e + (size_t)(row0 + rr) * kD + k0 + q * 4);
        Pack4 ph, pl;
        split4(x, ph, pl);
        *(uint2*)(e_hi + rr * CPAD + q * 4) = ph.u;
        *(uint2*)(e_lo + rr * CPAD + q * 4) = pl.u;
      }

      // ---- convert c chunk: LDS f32 (TDM-staged) -> LDS bf16 hi/lo ----
      const float* cbuf = c_f32 + (size_t)(k & 1) * CF32_FLOATS;
#pragma unroll 4
      for (int i = 0; i < 16; ++i) {
        int idx = tid + i * kBlock;          // 0..4095
        int vr  = idx >> 5;                  // vocab row 0..127
        int q   = idx & 31;
        float4 x = *(const float4*)(cbuf + vr * KCHUNK + q * 4);
        Pack4 ph, pl;
        split4(x, ph, pl);
        *(uint2*)(c_hi + vr * CPAD + q * 4) = ph.u;
        *(uint2*)(c_lo + vr * CPAD + q * 4) = pl.u;
      }
      __syncthreads();

      // ---- WMMA inner loop over the staged K chunk ----
      // B layout (32x16 bf16): lane = column, K(j<8)=8*kh+j, K(j>=8)=16+8*kh+j-8.
      const __bf16* bhbase = c_hi + (w * 16 + lane15) * CPAD + 8 * kh;
      const __bf16* blbase = c_lo + (w * 16 + lane15) * CPAD + 8 * kh;
#pragma unroll
      for (int kk = 0; kk < KCHUNK; kk += 32) {
        FragV bh, bl;
        bh.q[0] = *(const uint4*)(bhbase + kk);
        bh.q[1] = *(const uint4*)(bhbase + kk + 16);
        bl.q[0] = *(const uint4*)(blbase + kk);
        bl.q[1] = *(const uint4*)(blbase + kk + 16);
#pragma unroll
        for (int mt = 0; mt < MTILES; ++mt) {
          // A layout (16x32 bf16): lane = row, K(j<8)=8*kh+j, K(j>=8)=16+8*kh+j-8.
          const __bf16* ahp = e_hi + (mt * 16 + lane15) * CPAD + 8 * kh + kk;
          const __bf16* alp = e_lo + (mt * 16 + lane15) * CPAD + 8 * kh + kk;
          FragV ah, al;
          ah.q[0] = *(const uint4*)(ahp);
          ah.q[1] = *(const uint4*)(ahp + 16);
          al.q[0] = *(const uint4*)(alp);
          al.q[1] = *(const uint4*)(alp + 16);
          // bf16x3: hi*hi + hi*lo + lo*hi, f32 accumulate
          acc[mt] = __builtin_amdgcn_wmma_f32_16x16x32_bf16(
              false, ah.v, false, bh.v, (short)0, acc[mt], false, false);
          acc[mt] = __builtin_amdgcn_wmma_f32_16x16x32_bf16(
              false, ah.v, false, bl.v, (short)0, acc[mt], false, false);
          acc[mt] = __builtin_amdgcn_wmma_f32_16x16x32_bf16(
              false, al.v, false, bh.v, (short)0, acc[mt], false, false);
        }
      }
    }

    // ---- epilogue for this vocab chunk: bias, target capture, online softmax ----
    const int colg = v0 + w * 16 + lane15;
    const float bval = (colg < kV) ? bias[colg] : 0.f;
#pragma unroll
    for (int mt = 0; mt < MTILES; ++mt) {
#pragma unroll
      for (int r = 0; r < 8; ++r) {
        float val = acc[mt][r] + bval;
        if (colg >= kV) val = -__builtin_inff();
        if (colg == tgt_reg[mt][r]) t_logit[mt * 16 + kh * 8 + r] = val;
        float tm = val;
#pragma unroll
        for (int o = 1; o < 16; o <<= 1) tm = fmaxf(tm, __shfl_xor(tm, o, 16));
        float p = (tm > -3.0e38f) ? __expf(val - tm) : 0.f;
        float ts = p;
#pragma unroll
        for (int o = 1; o < 16; o <<= 1) ts += __shfl_xor(ts, o, 16);
        if (tm > -3.0e38f) {
          float mo = m_run[mt][r];
          float mn = fmaxf(mo, tm);
          s_run[mt][r] = s_run[mt][r] * __expf(mo - mn) + ts * __expf(tm - mn);
          m_run[mt][r] = mn;
        }
      }
    }
  }

  // ---- merge per-wave softmax state across the 8 waves ----
  if (lane15 == 0) {   // lanes 0 and 16 hold the reduced values for their rows
#pragma unroll
    for (int mt = 0; mt < MTILES; ++mt)
#pragma unroll
      for (int r = 0; r < 8; ++r) {
        int row = mt * 16 + kh * 8 + r;
        M_ws[w * MROWS + row] = m_run[mt][r];
        S_ws[w * MROWS + row] = s_run[mt][r];
      }
  }
  __syncthreads();

  if (tid < MROWS) {   // wave 0 finalizes all 32 rows
    const int row = tid;
    float m = -__builtin_inff(), s = 0.f;
#pragma unroll
    for (int ww = 0; ww < kWaves; ++ww) {
      float mw = M_ws[ww * MROWS + row];
      float sw = S_ws[ww * MROWS + row];
      if (mw > -3.0e38f) {
        float mn = fmaxf(m, mw);
        s = s * __expf(m - mn) + sw * __expf(mw - mn);
        m = mn;
      }
    }
    float lse   = m + __logf(s);
    int   t     = tgt_lds[row];
    bool  valid = (t != IGNORE);
    float nll   = valid ? (lse - t_logit[row]) : 0.f;
    float ns    = nll;
    float cv    = valid ? 1.f : 0.f;
#pragma unroll
    for (int o = 1; o < 32; o <<= 1) {
      ns += __shfl_xor(ns, o, 32);
      cv += __shfl_xor(cv, o, 32);
    }
    if (tid == 0) {
      atomicAdd(&gacc[0], ns);
      atomicAdd(&gacc[1], cv);
    }
  }
}

extern "C" void kernel_launch(void* const* d_in, const int* in_sizes, int n_in,
                              void* d_out, int out_size, void* d_ws, size_t ws_size,
                              hipStream_t stream) {
  (void)in_sizes; (void)n_in; (void)out_size; (void)ws_size;
  const float* e       = (const float*)d_in[0];
  const float* cmat    = (const float*)d_in[1];
  const int*   targets = (const int*)d_in[2];   // harness passes integer inputs as int32
  const float* bias    = (const float*)d_in[3];
  float*       out     = (float*)d_out;
  float*       acc     = (float*)d_ws;          // [0]=sum(nll), [1]=valid count

  (void)hipFuncSetAttribute((const void*)lce_main_kernel,
                            hipFuncAttributeMaxDynamicSharedMemorySize,
                            (int)LDS_BYTES);

  lce_init_kernel<<<1, 1, 0, stream>>>(acc);
  lce_main_kernel<<<dim3(kN / MROWS), dim3(kBlock), LDS_BYTES, stream>>>(
      e, cmat, targets, bias, acc);
  lce_fin_kernel<<<1, 1, 0, stream>>>(acc, out);
}
